// EquivariantBlock_16415365005677
// MI455X (gfx1250) — compile-verified, matrix operand order checked by reading, not run on hardware
//
#include <hip/hip_runtime.h>
#include <hip/hip_bf16.h>

#define Hdim 128
#define KE   288        // 2H + EF(=2), padded to multiple of 32
#define NT   8          // 128 / 16 output tiles
#define WAVES 4         // waves per block (blockDim = 128)

typedef __attribute__((ext_vector_type(16))) _Float16 v16h;
typedef __attribute__((ext_vector_type(8)))  _Float16 v8h;
typedef __attribute__((ext_vector_type(8)))  float    v8f;

__device__ inline v8f v8f_zero() {
  v8f z = {0.f,0.f,0.f,0.f,0.f,0.f,0.f,0.f};
  return z;
}

// A-fragment (16x32 f16) from row-major [16][lda] (LDS or global).
// ISA 16-bit A layout: lanes 0-15 -> M=lane, K base 0; lanes 16-31 -> M=lane-16, +8.
// VGPR v<4 -> K=2v,2v+1 ; v>=4 -> K=16+2(v-4),+1  (two contiguous 16B chunks per lane).
__device__ inline v16h load_a_frag(const _Float16* __restrict__ base, int lda, int kt, int lane) {
  const _Float16* r = base + (lane & 15) * lda + kt * 32 + ((lane & 16) ? 8 : 0);
  v16h a;
#pragma unroll
  for (int v = 0; v < 8; ++v) {
    int k = (v < 4) ? (v << 1) : (16 + ((v - 4) << 1));
    a[2*v]   = r[k];
    a[2*v+1] = r[k+1];
  }
  return a;
}

// Explicit register-resident weights: KT*8 B-fragments, one 32B chunk per lane each.
template<int KT>
__device__ inline void load_wfrags(const _Float16* __restrict__ pW, v16h* w, int lane) {
  const v16h* bW = (const v16h*)pW;
#pragma unroll
  for (int f = 0; f < KT * NT; ++f) w[f] = bW[f * 32 + lane];
}

template<int KT>
__device__ inline void gemm_rr(const _Float16* __restrict__ aBase, int lda,
                               const v16h* w, v8f acc[NT], int lane) {
#pragma unroll
  for (int kt = 0; kt < KT; ++kt) {
    v16h a = load_a_frag(aBase, lda, kt, lane);
#pragma unroll
    for (int nt = 0; nt < NT; ++nt)
      acc[nt] = __builtin_amdgcn_wmma_f32_16x16x32_f16(
          false, a, false, w[kt*NT + nt], (short)0, acc[nt], false, false);
  }
}

__device__ inline void add_bias(v8f acc[NT], const float* __restrict__ b, int lane) {
  int nl = lane & 15;
#pragma unroll
  for (int t = 0; t < NT; ++t) {
    float bb = b[t*16 + nl];
#pragma unroll
    for (int r = 0; r < 8; ++r) acc[t][r] += bb;
  }
}

__device__ inline float silu_f(float v) { return v / (1.0f + __expf(-v)); }

// LayerNorm over 128 features of each of 16 rows, then SiLU.
// Row m's features sit in VGPR index (m&7) across 8 tiles within one 16-lane half.
__device__ inline void layernorm_silu(v8f acc[NT], const float* __restrict__ g,
                                      const float* __restrict__ b, int lane) {
  int nl = lane & 15;
  float gv[NT], bv[NT];
#pragma unroll
  for (int t = 0; t < NT; ++t) { gv[t] = g[t*16+nl]; bv[t] = b[t*16+nl]; }
#pragma unroll
  for (int r = 0; r < 8; ++r) {
    float s = 0.f, s2 = 0.f;
#pragma unroll
    for (int t = 0; t < NT; ++t) { float v = acc[t][r]; s += v; s2 += v*v; }
#pragma unroll
    for (int off = 1; off < 16; off <<= 1) {   // xor<16 stays within a 16-lane half
      s  += __shfl_xor(s,  off, 32);
      s2 += __shfl_xor(s2, off, 32);
    }
    float mean = s * (1.f/128.f);
    float var  = s2 * (1.f/128.f) - mean*mean;
    float rstd = rsqrtf(var + 1e-5f);
#pragma unroll
    for (int t = 0; t < NT; ++t) {
      float v = (acc[t][r] - mean) * rstd * gv[t] + bv[t];
      acc[t][r] = silu_f(v);
    }
  }
}

// out[r] = dot(row_r, w[0..127]) for the 8 rows in this lane-half
__device__ inline void row_dot(const v8f acc[NT], const float* __restrict__ w,
                               int lane, float out[8]) {
  int nl = lane & 15;
  float wv[NT];
#pragma unroll
  for (int t = 0; t < NT; ++t) wv[t] = w[t*16+nl];
#pragma unroll
  for (int r = 0; r < 8; ++r) {
    float s = 0.f;
#pragma unroll
    for (int t = 0; t < NT; ++t) s += acc[t][r] * wv[t];
#pragma unroll
    for (int off = 1; off < 16; off <<= 1) s += __shfl_xor(s, off, 32);
    out[r] = s;
  }
}

// Write 16x128 accumulators as f16 rows into mid[rowBase..rowBase+15]
__device__ inline void store_rows_f16(_Float16* __restrict__ mid, int rowBase,
                                      const v8f acc[NT], int lane) {
  int nl = lane & 15, m0 = (lane & 16) ? 8 : 0;
#pragma unroll
  for (int t = 0; t < NT; ++t)
#pragma unroll
    for (int r = 0; r < 8; ++r)
      mid[(size_t)(rowBase + m0 + r) * Hdim + t*16 + nl] = (_Float16)acc[t][r];
}

// ---------------- prep kernels ----------------

// Pack f32 weight [din x dout] into f16 B-fragment order:
// dst[((kt*NT+nt)*32+lane)*16+i] = W[k][n], n=nt*16+(lane&15), k per ISA B layout.
__global__ void pack_w_kernel(const float* __restrict__ src, _Float16* __restrict__ dst,
                              int din, int dout, int KT) {
  int idx = blockIdx.x * blockDim.x + threadIdx.x;
  int nT = dout >> 4;
  int total = KT * nT * 32 * 16;
  if (idx >= total) return;
  int i    = idx & 15;
  int lane = (idx >> 4) & 31;
  int fr   = idx >> 9;
  int nt   = fr % nT, kt = fr / nT;
  int v = i >> 1, p = i & 1;
  int k = kt*32 + ((v < 4) ? (v << 1) : (16 + ((v - 4) << 1))) + ((lane & 16) ? 8 : 0) + p;
  int n = nt*16 + (lane & 15);
  float val = (k < din) ? src[k * dout + n] : 0.0f;
  dst[idx] = (_Float16)val;
}

__global__ void h_init_kernel(const float* __restrict__ h, float* __restrict__ hcur,
                              _Float16* __restrict__ h16, int n) {
  int i = blockIdx.x * blockDim.x + threadIdx.x;
  if (i >= n) return;
  float v = h[i];
  hcur[i] = v;
  h16[i] = (_Float16)v;
}

__global__ void zero_f32_kernel(float* __restrict__ p, int n) {
  int i = blockIdx.x * blockDim.x + threadIdx.x;
  if (i < n) p[i] = 0.0f;
}

// coord2diff: cd (normalized), ea16 = [radial, edge_attr] as f16
__global__ void edge_pre_kernel(const float* __restrict__ x, const int* __restrict__ ei,
                                const float* __restrict__ eattr,
                                float* __restrict__ cd, _Float16* __restrict__ ea16, int nE) {
  int e = blockIdx.x * blockDim.x + threadIdx.x;
  if (e >= nE) return;
  int r = ei[e], c = ei[nE + e];
  float d0 = x[r*3+0] - x[c*3+0];
  float d1 = x[r*3+1] - x[c*3+1];
  float d2 = x[r*3+2] - x[c*3+2];
  float radial = d0*d0 + d1*d1 + d2*d2;
  float inv = 1.0f / (sqrtf(radial + 1e-8f) + 1.0f);   // NORM_CONST = 1.0
  cd[e*3+0] = d0*inv; cd[e*3+1] = d1*inv; cd[e*3+2] = d2*inv;
  ea16[e*2+0] = (_Float16)radial;
  ea16[e*2+1] = (_Float16)eattr[e];
}

__global__ void finalize_kernel(const float* __restrict__ hcur, const float* __restrict__ x,
                                const float* __restrict__ xagg, const float* __restrict__ nmask,
                                float* __restrict__ out, int nN) {
  int i = blockIdx.x * blockDim.x + threadIdx.x;
  int nh = nN * Hdim;
  if (i < nh) {
    out[i] = hcur[i] * nmask[i / Hdim];
  } else if (i < nh + nN * 3) {
    int j = i - nh;
    out[i] = (x[j] + xagg[j]) * nmask[j / 3];   // xagg already includes /NORM_FACTOR
  }
}

// ---------------- pass-1 kernels (weight-stationary, K->128, LN+SiLU, write mid) ----------------

// Edge/equiv layer-1: gather [h[row]|h[col]|radial,attr] -> [16,288]x[288,128]
__global__ __launch_bounds__(128, 1) void mlp1_gather_kernel(
    const _Float16* __restrict__ h16, const _Float16* __restrict__ ea16,
    const int* __restrict__ ei,
    const _Float16* __restrict__ pW1, const float* __restrict__ b1,
    const float* __restrict__ g1, const float* __restrict__ bb1,
    _Float16* __restrict__ mid, int eBase, int nChunk, int nE) {
  __shared__ _Float16 smem[WAVES * 16 * KE];
  const int lane = threadIdx.x & 31;
  const int wave = threadIdx.x >> 5;
  _Float16* aT = smem + wave * 16 * KE;
  v16h w[9 * NT];
  load_wfrags<9>(pW1, w, lane);           // 576 VGPRs, register-resident
  // one-time zero of pad columns 258..287 (staging never writes them)
  for (int idx = lane; idx < 16 * 15; idx += 32) {
    int e = idx / 15, c = idx % 15;
    *(unsigned int*)(aT + e * KE + 258 + c * 2) = 0u;
  }
  const int half = lane >> 4;             // 0: h[row] half, 1: h[col] half
  const int ch   = lane & 15;             // 16B chunk within the 256B row
  const int ntiles = nChunk >> 4;
  const int stride = (int)gridDim.x * WAVES;
  for (int tile = blockIdx.x * WAVES + wave; tile < ntiles; tile += stride) {
    const int e0 = eBase + (tile << 4);
    // vectorized gather: one b128 global load + one b128 LDS store per lane per edge
    for (int e = 0; e < 16; ++e) {
      const int er = ei[e0 + e], ec = ei[nE + e0 + e];
      const _Float16* src = h16 + (half ? ec : er) * Hdim + ch * 8;
      v8h vv = *(const v8h*)src;
      *(v8h*)(aT + e * KE + half * Hdim + ch * 8) = vv;
    }
    if (lane < 16)                        // [radial, edge_attr] tail, one edge per lane
      *(unsigned int*)(aT + lane * KE + 2 * Hdim) =
          *(const unsigned int*)(ea16 + (size_t)(e0 + lane) * 2);
    // prefetch next tile's gathered rows so the random gather overlaps the WMMA chain
    int tn = tile + stride;
    if (tn < ntiles && lane < 16) {
      int gidx = eBase + (tn << 4) + lane;
      __builtin_prefetch(h16 + ei[gidx] * Hdim, 0, 0);
      __builtin_prefetch(h16 + ei[nE + gidx] * Hdim, 0, 0);
    }
    v8f acc[NT];
#pragma unroll
    for (int t = 0; t < NT; ++t) acc[t] = v8f_zero();
    gemm_rr<9>(aT, KE, w, acc, lane);
    add_bias(acc, b1, lane);
    layernorm_silu(acc, g1, bb1, lane);
    store_rows_f16(mid, tile << 4, acc, lane);    // mid indexed by chunk-local row
  }
}

// Node layer-1: [h|agg] -> [16,256]x[256,128]
__global__ __launch_bounds__(128, 1) void node1_kernel(
    const _Float16* __restrict__ h16, const float* __restrict__ aggv,
    const _Float16* __restrict__ pW1, const float* __restrict__ b1,
    const float* __restrict__ g1, const float* __restrict__ bb1,
    _Float16* __restrict__ mid, int nN) {
  __shared__ _Float16 smem[WAVES * 16 * 256];
  const int lane = threadIdx.x & 31;
  const int wave = threadIdx.x >> 5;
  _Float16* aT = smem + wave * 16 * 256;
  v16h w[8 * NT];
  load_wfrags<8>(pW1, w, lane);           // 512 VGPRs
  const int half = lane >> 4, ch = lane & 15;
  const int ntiles = nN >> 4;
  const int stride = (int)gridDim.x * WAVES;
  for (int tile = blockIdx.x * WAVES + wave; tile < ntiles; tile += stride) {
    const int n0 = tile << 4;
    for (int i = 0; i < 16; ++i) {
      int node = n0 + i;
      if (half == 0) {                    // h16 half: straight 16B copy
        v8h vv = *(const v8h*)(h16 + node * Hdim + ch * 8);
        *(v8h*)(aT + i * 256 + ch * 8) = vv;
      } else {                            // agg half: 8 f32 -> 8 f16, one 16B store
        float4 f0 = *(const float4*)(aggv + node * Hdim + ch * 8);
        float4 f1 = *(const float4*)(aggv + node * Hdim + ch * 8 + 4);
        v8h t;
        t[0] = (_Float16)f0.x; t[1] = (_Float16)f0.y;
        t[2] = (_Float16)f0.z; t[3] = (_Float16)f0.w;
        t[4] = (_Float16)f1.x; t[5] = (_Float16)f1.y;
        t[6] = (_Float16)f1.z; t[7] = (_Float16)f1.w;
        *(v8h*)(aT + i * 256 + Hdim + ch * 8) = t;
      }
    }
    v8f acc[NT];
#pragma unroll
    for (int t = 0; t < NT; ++t) acc[t] = v8f_zero();
    gemm_rr<8>(aT, 256, w, acc, lane);
    add_bias(acc, b1, lane);
    layernorm_silu(acc, g1, bb1, lane);
    store_rows_f16(mid, n0, acc, lane);
  }
}

// ---------------- pass-2 kernels (K=128, A-fragments straight from global mid) ----------------

// Edge layer-2: SiLU -> att sigmoid -> scaled atomic segment-sum into agg
__global__ __launch_bounds__(128, 1) void edge2_kernel(
    const _Float16* __restrict__ mid, const int* __restrict__ ei,
    const float* __restrict__ emask,
    const _Float16* __restrict__ pW2, const float* __restrict__ b2,
    const float* __restrict__ attw, const float* __restrict__ attb,
    float* __restrict__ agg, int eBase, int nChunk) {
  const int lane = threadIdx.x & 31;
  const int wave = threadIdx.x >> 5;
  v16h w[4 * NT];
  load_wfrags<4>(pW2, w, lane);           // 256 VGPRs
  const int ntiles = nChunk >> 4;
  const int stride = (int)gridDim.x * WAVES;
  for (int tile = blockIdx.x * WAVES + wave; tile < ntiles; tile += stride) {
    v8f acc[NT];
#pragma unroll
    for (int t = 0; t < NT; ++t) acc[t] = v8f_zero();
    gemm_rr<4>(mid + (size_t)(tile << 4) * Hdim, Hdim, w, acc, lane);
    add_bias(acc, b2, lane);
#pragma unroll
    for (int t = 0; t < NT; ++t)
#pragma unroll
      for (int r = 0; r < 8; ++r) acc[t][r] = silu_f(acc[t][r]);
    float attv[8];
    row_dot(acc, attw, lane, attv);
    const float ab = attb[0];
    const int m0 = (lane & 16) ? 8 : 0, nl = lane & 15;
    const int e0 = eBase + (tile << 4);
    float scl[8]; int rw[8];
#pragma unroll
    for (int r = 0; r < 8; ++r) {
      int e = e0 + m0 + r;
      rw[r] = ei[e];
      float a = 1.f / (1.f + __expf(-(attv[r] + ab)));
      scl[r] = a * emask[e] * 0.01f;      // fold /NORM_FACTOR
    }
#pragma unroll
    for (int t = 0; t < NT; ++t)
#pragma unroll
      for (int r = 0; r < 8; ++r)
        atomicAdd(&agg[rw[r] * Hdim + t*16 + nl], acc[t][r] * scl[r]);
  }
}

// Node layer-2: residual + mask, update hcur & h16
__global__ __launch_bounds__(128, 1) void node2_kernel(
    const _Float16* __restrict__ mid, float* __restrict__ hcur,
    _Float16* __restrict__ h16,
    const _Float16* __restrict__ pW2, const float* __restrict__ b2,
    const float* __restrict__ nmask, int nN) {
  const int lane = threadIdx.x & 31;
  const int wave = threadIdx.x >> 5;
  v16h w[4 * NT];
  load_wfrags<4>(pW2, w, lane);
  const int ntiles = nN >> 4;
  const int stride = (int)gridDim.x * WAVES;
  for (int tile = blockIdx.x * WAVES + wave; tile < ntiles; tile += stride) {
    v8f acc[NT];
#pragma unroll
    for (int t = 0; t < NT; ++t) acc[t] = v8f_zero();
    gemm_rr<4>(mid + (size_t)(tile << 4) * Hdim, Hdim, w, acc, lane);
    add_bias(acc, b2, lane);
    const int m0 = (lane & 16) ? 8 : 0, nl = lane & 15;
    const int n0 = tile << 4;
#pragma unroll
    for (int r = 0; r < 8; ++r) {
      int node = n0 + m0 + r;
      float nm = nmask[node];
#pragma unroll
      for (int t = 0; t < NT; ++t) {
        int idx = node*Hdim + t*16 + nl;
        float v = (hcur[idx] + acc[t][r]) * nm;
        hcur[idx] = v;
        h16[idx] = (_Float16)v;
      }
    }
  }
}

// Equivariant layer-2: LN2+SiLU -> dot(w3) -> cd scale -> atomic segment-sum into x_agg
__global__ __launch_bounds__(128, 1) void equiv2_kernel(
    const _Float16* __restrict__ mid, const int* __restrict__ ei,
    const float* __restrict__ emask, const float* __restrict__ cd,
    const _Float16* __restrict__ pW2, const float* __restrict__ b2,
    const float* __restrict__ g2, const float* __restrict__ bb2,
    const float* __restrict__ w3,
    float* __restrict__ xagg, int eBase, int nChunk) {
  const int lane = threadIdx.x & 31;
  const int wave = threadIdx.x >> 5;
  v16h w[4 * NT];
  load_wfrags<4>(pW2, w, lane);
  const int ntiles = nChunk >> 4;
  const int stride = (int)gridDim.x * WAVES;
  for (int tile = blockIdx.x * WAVES + wave; tile < ntiles; tile += stride) {
    v8f acc[NT];
#pragma unroll
    for (int t = 0; t < NT; ++t) acc[t] = v8f_zero();
    gemm_rr<4>(mid + (size_t)(tile << 4) * Hdim, Hdim, w, acc, lane);
    add_bias(acc, b2, lane);
    layernorm_silu(acc, g2, bb2, lane);
    float phi[8];
    row_dot(acc, w3, lane, phi);
    const int m0 = (lane & 16) ? 8 : 0, nl = lane & 15;
    const int e0 = eBase + (tile << 4);
    if (nl == 0) {                        // lane 0 / lane 16 cover the two row halves
#pragma unroll
      for (int r = 0; r < 8; ++r) {
        int e = e0 + m0 + r;
        float p = phi[r] * emask[e] * 0.01f;      // fold /NORM_FACTOR
        int row = ei[e];
        atomicAdd(&xagg[row*3+0], cd[e*3+0] * p);
        atomicAdd(&xagg[row*3+1], cd[e*3+1] * p);
        atomicAdd(&xagg[row*3+2], cd[e*3+2] * p);
      }
    }
  }
}

// ---------------- host launch ----------------

extern "C" void kernel_launch(void* const* d_in, const int* in_sizes, int n_in,
                              void* d_out, int out_size, void* d_ws, size_t ws_size,
                              hipStream_t stream) {
  (void)n_in; (void)out_size;
  const float* h     = (const float*)d_in[0];
  const float* x     = (const float*)d_in[1];
  const int*   ei    = (const int*)d_in[2];
  const float* eattr = (const float*)d_in[3];
  const float* nmask = (const float*)d_in[4];
  const float* emask = (const float*)d_in[5];
  const int nN = in_sizes[0] / Hdim;
  const int nE = in_sizes[2] / 2;

  // params: jax pytree order (sorted dict keys): equiv, gcl_0, gcl_1
  const float* eq_l1b  = (const float*)d_in[6];
  const float* eq_l1w  = (const float*)d_in[7];
  const float* eq_l2b  = (const float*)d_in[8];
  const float* eq_l2w  = (const float*)d_in[9];
  const float* eq_l3w  = (const float*)d_in[10];
  const float* eq_ln1b = (const float*)d_in[11];
  const float* eq_ln1g = (const float*)d_in[12];
  const float* eq_ln2b = (const float*)d_in[13];
  const float* eq_ln2g = (const float*)d_in[14];
  const float *att_b[2], *att_w[2], *e_l1b[2], *e_l1w[2], *e_l2b[2], *e_l2w[2],
              *e_ln1b[2], *e_ln1g[2], *n_l1b[2], *n_l1w[2], *n_l2b[2], *n_l2w[2],
              *n_ln1b[2], *n_ln1g[2];
  for (int i = 0; i < 2; ++i) {
    int b = 15 + 14*i;
    att_b[i]  = (const float*)d_in[b+0];  att_w[i]  = (const float*)d_in[b+1];
    e_l1b[i]  = (const float*)d_in[b+2];  e_l1w[i]  = (const float*)d_in[b+3];
    e_l2b[i]  = (const float*)d_in[b+4];  e_l2w[i]  = (const float*)d_in[b+5];
    e_ln1b[i] = (const float*)d_in[b+6];  e_ln1g[i] = (const float*)d_in[b+7];
    n_l1b[i]  = (const float*)d_in[b+8];  n_l1w[i]  = (const float*)d_in[b+9];
    n_l2b[i]  = (const float*)d_in[b+10]; n_l2w[i]  = (const float*)d_in[b+11];
    n_ln1b[i] = (const float*)d_in[b+12]; n_ln1g[i] = (const float*)d_in[b+13];
  }

  // workspace bump allocator (256B aligned)
  char* wsb = (char*)d_ws;
  size_t off = 0;
  auto alloc = [&](size_t bytes) -> void* {
    void* p = wsb + off;
    off = (off + bytes + 255) & ~(size_t)255;
    return p;
  };
  float*     hcur = (float*)alloc((size_t)nN * Hdim * 4);
  float*     agg  = (float*)alloc((size_t)nN * Hdim * 4);
  float*     xagg = (float*)alloc((size_t)nN * 3 * 4);
  float*     cd   = (float*)alloc((size_t)nE * 3 * 4);
  _Float16*  h16  = (_Float16*)alloc((size_t)nN * Hdim * 2);
  _Float16*  ea16 = (_Float16*)alloc((size_t)nE * 2 * 2);
  _Float16 *pEdge1[2], *pEdge2[2], *pNode1[2], *pNode2[2];
  for (int i = 0; i < 2; ++i) {
    pEdge1[i] = (_Float16*)alloc((size_t)KE  * Hdim * 2);
    pEdge2[i] = (_Float16*)alloc((size_t)128 * Hdim * 2);
    pNode1[i] = (_Float16*)alloc((size_t)256 * Hdim * 2);
    pNode2[i] = (_Float16*)alloc((size_t)128 * Hdim * 2);
  }
  _Float16* pEq1 = (_Float16*)alloc((size_t)KE  * Hdim * 2);
  _Float16* pEq2 = (_Float16*)alloc((size_t)128 * Hdim * 2);

  // mid buffer: sized from remaining workspace; edge passes chunked over it
  size_t avail = (ws_size > off) ? (ws_size - off) : ((size_t)nN * Hdim * 2 + 4096);
  size_t rows  = avail / (Hdim * sizeof(_Float16));
  int chunkE = (rows >= (size_t)nE) ? nE : (int)(rows & ~(size_t)15);
  if (chunkE < 16) chunkE = 16;
  int midRows = (chunkE > nN) ? chunkE : nN;
  _Float16* mid = (_Float16*)alloc((size_t)midRows * Hdim * 2);

  auto cdiv = [](int a, int b) { return (a + b - 1) / b; };
  auto pack = [&](const float* src, _Float16* dst, int din) {
    int KT = (din + 31) / 32;
    int total = KT * NT * 32 * 16;
    pack_w_kernel<<<cdiv(total, 256), 256, 0, stream>>>(src, dst, din, Hdim, KT);
  };
  for (int i = 0; i < 2; ++i) {
    pack(e_l1w[i], pEdge1[i], 258);
    pack(e_l2w[i], pEdge2[i], 128);
    pack(n_l1w[i], pNode1[i], 256);
    pack(n_l2w[i], pNode2[i], 128);
  }
  pack(eq_l1w, pEq1, 258);
  pack(eq_l2w, pEq2, 128);

  h_init_kernel<<<cdiv(nN * Hdim, 256), 256, 0, stream>>>(h, hcur, h16, nN * Hdim);
  edge_pre_kernel<<<cdiv(nE, 256), 256, 0, stream>>>(x, ei, eattr, cd, ea16, nE);

  const int ntilesN = nN >> 4;
  const int nb1 = (cdiv(ntilesN, WAVES * 2) > 0) ? cdiv(ntilesN, WAVES * 2) : 1;

  for (int i = 0; i < 2; ++i) {
    zero_f32_kernel<<<cdiv(nN * Hdim, 256), 256, 0, stream>>>(agg, nN * Hdim);
    for (int eBase = 0; eBase < nE; eBase += chunkE) {
      int ce = (nE - eBase < chunkE) ? (nE - eBase) : chunkE;
      int nt = ce >> 4;
      int g1 = cdiv(nt, WAVES * 8); if (g1 < 1) g1 = 1;   // ~8 tiles per wave
      int g2 = cdiv(nt, WAVES * 4); if (g2 < 1) g2 = 1;
      mlp1_gather_kernel<<<g1, 128, 0, stream>>>(
          h16, ea16, ei, pEdge1[i], e_l1b[i], e_ln1g[i], e_ln1b[i],
          mid, eBase, ce, nE);
      edge2_kernel<<<g2, 128, 0, stream>>>(
          mid, ei, emask, pEdge2[i], e_l2b[i], att_w[i], att_b[i],
          agg, eBase, ce);
    }
    node1_kernel<<<nb1, 128, 0, stream>>>(
        h16, agg, pNode1[i], n_l1b[i], n_ln1g[i], n_ln1b[i], mid, nN);
    node2_kernel<<<nb1, 128, 0, stream>>>(
        mid, hcur, h16, pNode2[i], n_l2b[i], nmask, nN);
  }

  zero_f32_kernel<<<cdiv(nN * 3, 256), 256, 0, stream>>>(xagg, nN * 3);
  for (int eBase = 0; eBase < nE; eBase += chunkE) {
    int ce = (nE - eBase < chunkE) ? (nE - eBase) : chunkE;
    int nt = ce >> 4;
    int g1 = cdiv(nt, WAVES * 8); if (g1 < 1) g1 = 1;
    int g2 = cdiv(nt, WAVES * 4); if (g2 < 1) g2 = 1;
    mlp1_gather_kernel<<<g1, 128, 0, stream>>>(
        h16, ea16, ei, pEq1, eq_l1b, eq_ln1g, eq_ln1b, mid, eBase, ce, nE);
    equiv2_kernel<<<g2, 128, 0, stream>>>(
        mid, ei, emask, cd, pEq2, eq_l2b, eq_ln2g, eq_ln2b, eq_l3w,
        xagg, eBase, ce);
  }

  int total = nN * Hdim + nN * 3;
  finalize_kernel<<<cdiv(total, 256), 256, 0, stream>>>(hcur, x, xagg, nmask,
                                                        (float*)d_out, nN);
}
